// Rs_GCN_29137058136127
// MI455X (gfx1250) — matrix operand order repaired and checked by reading, not compiled, not run on hardware
//
#include <hip/hip_runtime.h>
#include <stdint.h>

#define BB    8
#define NN    4096
#define CC    1024
#define ICN   256
#define ROWS  (BB * NN)          // 32768
#define BN_EPS 1e-5f

typedef __attribute__((ext_vector_type(16))) __bf16 v16bf;
typedef __attribute__((ext_vector_type(8)))  float  v8f;

union Frag { v16bf v; uint4 q[2]; };

// ---------- bf16 helpers (round-to-nearest-even) ----------
__device__ __forceinline__ uint16_t f2bf(float x) {
    uint32_t u = __float_as_uint(x);
    uint32_t r = u + 0x7FFFu + ((u >> 16) & 1u);
    return (uint16_t)(r >> 16);
}
__device__ __forceinline__ uint32_t pack2(float a, float b) {
    return (uint32_t)f2bf(a) | ((uint32_t)f2bf(b) << 16);
}

// ---------- WMMA wrapper ----------
__device__ __forceinline__ v8f wmma_bf16(v16bf a, v16bf b, v8f c) {
    return __builtin_amdgcn_wmma_f32_16x16x32_bf16(
        /*neg_a=*/false, a, /*neg_b=*/false, b,
        /*c_mod=*/(short)0, c, /*reuse_a=*/false, /*reuse_b=*/false);
}

// ---------- CDNA5 async global->LDS copy (ASYNCcnt tracked, bypasses VGPRs) ----------
__device__ __forceinline__ void async_o(uint32_t lds_off, const uint16_t* gsrc) {
    asm volatile("global_load_async_to_lds_b128 %0, %1, off"
                 :: "v"(lds_off), "v"(gsrc) : "memory");
}
__device__ __forceinline__ void wait_async0() {
    asm volatile("s_wait_asynccnt 0x0" ::: "memory");
}
__device__ __forceinline__ uint32_t lds_off(const uint16_t* p) {
    return (uint32_t)(uintptr_t)p;   // low 32 bits of flat shared addr = LDS offset
}

// ---------- LDS fragment loads (layouts per CDNA5 ISA 7.12.2) ----------
// A 16x32 bf16: lanes 0-15 rows M=lane, K {0..7,16..23}; lanes 16-31 K {8..15,24..31}
__device__ __forceinline__ v16bf load_frag_a(const uint16_t* sA, int lane, int m_off) {
    int row = m_off + (lane & 15);
    int k0  = (lane >> 4) * 8;
    Frag f;
    f.q[0] = *(const uint4*)(sA + row * 40 + k0);
    f.q[1] = *(const uint4*)(sA + row * 40 + k0 + 16);
    return f.v;
}
// B 32x16 bf16 staged as B^T row-major: lane col = lane&15; lanes 0-15 K {0..7,16..23}; lanes 16-31 K {8..15,24..31}
__device__ __forceinline__ v16bf load_frag_b(const uint16_t* sB, int lane, int n_off) {
    int col = n_off + (lane & 15);
    int k0  = (lane >> 4) * 8;
    Frag f;
    f.q[0] = *(const uint4*)(sB + col * 40 + k0);
    f.q[1] = *(const uint4*)(sB + col * 40 + k0 + 16);
    return f.v;
}

// wave tile 32x64: 2 A frags x 4 B frags -> 8 WMMA per K-step
__device__ __forceinline__ void mma_step(const uint16_t* sA, const uint16_t* sB,
                                         int lane, int m0, int n0, v8f acc[2][4]) {
    v16bf a0 = load_frag_a(sA, lane, m0);
    v16bf a1 = load_frag_a(sA, lane, m0 + 16);
    v16bf b0 = load_frag_b(sB, lane, n0);
    v16bf b1 = load_frag_b(sB, lane, n0 + 16);
    v16bf b2 = load_frag_b(sB, lane, n0 + 32);
    v16bf b3 = load_frag_b(sB, lane, n0 + 48);
    acc[0][0] = wmma_bf16(a0, b0, acc[0][0]);
    acc[0][1] = wmma_bf16(a0, b1, acc[0][1]);
    acc[0][2] = wmma_bf16(a0, b2, acc[0][2]);
    acc[0][3] = wmma_bf16(a0, b3, acc[0][3]);
    acc[1][0] = wmma_bf16(a1, b0, acc[1][0]);
    acc[1][1] = wmma_bf16(a1, b1, acc[1][1]);
    acc[1][2] = wmma_bf16(a1, b2, acc[1][2]);
    acc[1][3] = wmma_bf16(a1, b3, acc[1][3]);
}

// ---------- double-buffered WMMA GEMM mainloop (block tile 128x128, K % 64 == 0) ----
// Strength-reduced: 4 global src pointers advance by 64B per K-step; 8 loop-invariant
// LDS destination offsets (4 chunks x 2 buffers) precomputed before the loop.
__device__ __forceinline__ void gemm_loop(const uint16_t* Asrc, long lda, long rbase,
                                          const uint16_t* Bsrc, long ldb, long nbase,
                                          int K,
                                          uint16_t* sA0, uint16_t* sA1,
                                          uint16_t* sB0, uint16_t* sB1,
                                          int tid, int lane, int m0, int n0,
                                          v8f acc[2][4]) {
    int row = tid >> 2;               // [0,64): this thread's chunk rows are row, row+64
    int kq  = (tid & 3) * 8;          // 8-element (16B) chunk within the 32-wide K slice

    const uint16_t* gA0 = Asrc + (rbase + row) * lda + kq;
    const uint16_t* gA1 = gA0 + 64 * lda;
    const uint16_t* gB0 = Bsrc + (nbase + row) * ldb + kq;
    const uint16_t* gB1 = gB0 + 64 * ldb;

    const uint32_t oA0 = lds_off(sA0 + row * 40 + kq), oA0b = oA0 + 64 * 40 * 2;
    const uint32_t oA1 = lds_off(sA1 + row * 40 + kq), oA1b = oA1 + 64 * 40 * 2;
    const uint32_t oB0 = lds_off(sB0 + row * 40 + kq), oB0b = oB0 + 64 * 40 * 2;
    const uint32_t oB1 = lds_off(sB1 + row * 40 + kq), oB1b = oB1 + 64 * 40 * 2;

#define ISSUE_TILES(a_lo, a_hi, b_lo, b_hi)                       \
    {                                                             \
        async_o(a_lo, gA0); async_o(a_hi, gA1);                   \
        async_o(b_lo, gB0); async_o(b_hi, gB1);                   \
        gA0 += 32; gA1 += 32; gB0 += 32; gB1 += 32;               \
    }

    ISSUE_TILES(oA0, oA0b, oB0, oB0b);          // prime buffer 0 (k = 0)

    for (int k = 0; k < K; k += 64) {
        // ---- even K-step: compute buf0, prefetch buf1 ----
        wait_async0();            // my async issues into buf0 are complete
        __syncthreads();          // => everyone's issues complete; buf1 free
        if (k + 32 < K) ISSUE_TILES(oA1, oA1b, oB1, oB1b);
        mma_step(sA0, sB0, lane, m0, n0, acc);
        // ---- odd K-step: compute buf1, prefetch buf0 ----
        wait_async0();
        __syncthreads();
        if (k + 64 < K) ISSUE_TILES(oA0, oA0b, oB0, oB0b);
        mma_step(sA1, sB1, lane, m0, n0, acc);
    }
#undef ISSUE_TILES
}

// ---------- epilogue stores ----------
__device__ __forceinline__ void store_bf16_rowmajor(uint16_t* dst, long ld, long rbase,
                                                    long cbase, int lane,
                                                    const float* bias, v8f acc[2][4]) {
#pragma unroll
    for (int tm = 0; tm < 2; ++tm)
#pragma unroll
        for (int tn = 0; tn < 4; ++tn) {
            long col = cbase + tn * 16 + (lane & 15);
            float bv = bias ? bias[col] : 0.0f;
            long r0  = rbase + tm * 16 + 8 * (lane >> 4);
#pragma unroll
            for (int p = 0; p < 8; ++p)
                dst[(r0 + p) * ld + col] = f2bf(acc[tm][tn][p] + bv);
        }
}
// transposed bf16 store: dst[col * ldT + row]; one b128 store per lane per tile
__device__ __forceinline__ void store_bf16_transposed(uint16_t* dst, long ldT, long rbase,
                                                      long cbase, int lane, const float* bias,
                                                      float scale, v8f acc[2][4]) {
#pragma unroll
    for (int tm = 0; tm < 2; ++tm)
#pragma unroll
        for (int tn = 0; tn < 4; ++tn) {
            long col = cbase + tn * 16 + (lane & 15);
            float bv = bias ? bias[col] : 0.0f;
            long r0  = rbase + tm * 16 + 8 * (lane >> 4);
            uint4 q;
            uint32_t* u = (uint32_t*)&q;
#pragma unroll
            for (int i = 0; i < 4; ++i)
                u[i] = pack2((acc[tm][tn][2 * i] + bv) * scale,
                             (acc[tm][tn][2 * i + 1] + bv) * scale);
            *(uint4*)(dst + col * ldT + r0) = q;
        }
}

#define GEMM_PROLOG                                                     \
    __shared__ __align__(16) uint16_t sA0[128 * 40], sA1[128 * 40];     \
    __shared__ __align__(16) uint16_t sB0[128 * 40], sB1[128 * 40];     \
    int tid = threadIdx.x, lane = tid & 31, wave = tid >> 5;            \
    int m0 = (wave >> 1) * 32, n0 = (wave & 1) * 64;                    \
    v8f acc[2][4] = {};

// ================= Kernel 0: weight f32->bf16 + zero stats =================
__global__ void prep_kernel(const float* gw, const float* tw, const float* pw, const float* Ww,
                            uint16_t* gw_bf, uint16_t* tw_bf, uint16_t* pw_bf, uint16_t* Ww_bf,
                            float* stats) {
    int i = blockIdx.x * blockDim.x + threadIdx.x;
    const int NW = ICN * CC;
    if (i < NW) {
        gw_bf[i] = f2bf(gw[i]);
        tw_bf[i] = f2bf(tw[i]);
        pw_bf[i] = f2bf(pw[i]);
        Ww_bf[i] = f2bf(Ww[i]);
    }
    if (i < 2 * CC) stats[i] = 0.0f;
}

// ================= Kernel 0b: v f32 -> bf16 (one pass, off the GEMM hot path) ===========
__global__ void conv_v_kernel(const float* __restrict__ v, uint16_t* __restrict__ v_bf) {
    long i = ((long)blockIdx.x * blockDim.x + threadIdx.x) * 4;
    float4 d = *(const float4*)(v + i);
    uint2 p;
    p.x = pack2(d.x, d.y);
    p.y = pack2(d.z, d.w);
    *(uint2*)(v_bf + i) = p;
}

// ================= Kernel A: projections theta/phi/g =================
// z=0: theta -> row-major (ROWS, IC); z=1: phi -> (B, IC, N); z=2: g -> (B, IC, N)
__global__ void __launch_bounds__(256)
proj_kernel(const uint16_t* __restrict__ v_bf,
            const uint16_t* __restrict__ tw_bf, const uint16_t* __restrict__ pw_bf,
            const uint16_t* __restrict__ gw_bf,
            const float* __restrict__ tb, const float* __restrict__ pb,
            const float* __restrict__ gb,
            uint16_t* __restrict__ theta_bf, uint16_t* __restrict__ phiT,
            uint16_t* __restrict__ gT) {
    GEMM_PROLOG
    long rbase = (long)blockIdx.y * 128;
    long cbase = (long)blockIdx.x * 128;
    int which = blockIdx.z;
    const uint16_t* wgt = (which == 0) ? tw_bf : (which == 1) ? pw_bf : gw_bf;

    gemm_loop(v_bf, CC, rbase, wgt, CC, cbase, CC,
              sA0, sA1, sB0, sB1, tid, lane, m0, n0, acc);

    long wr = rbase + m0;
    long wc = cbase + n0;
    if (which == 0) {
        store_bf16_rowmajor(theta_bf, ICN, wr, wc, lane, tb, acc);
    } else {
        long b      = rbase >> 12;                 // block fully inside one batch (128 | 4096)
        long n_in_b = (rbase & (NN - 1)) + m0;
        uint16_t* dst    = ((which == 1) ? phiT : gT) + b * (long)ICN * NN;
        const float* bia = (which == 1) ? pb : gb;
        store_bf16_transposed(dst, NN, n_in_b, wc, lane, bia, 1.0f, acc);
    }
}

// ================= Kernel B: S_b = phi^T g  (store S^T / N) =================
__global__ void __launch_bounds__(256)
s_kernel(const uint16_t* __restrict__ phiT, const uint16_t* __restrict__ gT,
         uint16_t* __restrict__ sT) {
    GEMM_PROLOG
    long b     = blockIdx.z;
    long rbase = (long)blockIdx.y * 128;   // i
    long cbase = (long)blockIdx.x * 128;   // j
    const uint16_t* A  = phiT + b * (long)ICN * NN;
    const uint16_t* Bm = gT   + b * (long)ICN * NN;

    gemm_loop(A, NN, rbase, Bm, NN, cbase, NN,
              sA0, sA1, sB0, sB1, tid, lane, m0, n0, acc);

    // sT[b][j][i] = S[i][j] / N
    store_bf16_transposed(sT + b * (long)ICN * ICN, ICN, rbase + m0, cbase + n0,
                          lane, nullptr, 1.0f / (float)NN, acc);
}

// ================= Kernel C: y = theta @ (S/N) =================
__global__ void __launch_bounds__(256)
y_kernel(const uint16_t* __restrict__ theta_bf, const uint16_t* __restrict__ sT,
         uint16_t* __restrict__ y_bf) {
    GEMM_PROLOG
    long rbase = (long)blockIdx.y * 128;   // global row
    long cbase = (long)blockIdx.x * 128;   // i
    long b     = rbase >> 12;
    const uint16_t* Bm = sT + b * (long)ICN * ICN;

    gemm_loop(theta_bf, ICN, rbase, Bm, ICN, cbase, ICN,
              sA0, sA1, sB0, sB1, tid, lane, m0, n0, acc);

    store_bf16_rowmajor(y_bf, ICN, rbase + m0, cbase + n0, lane, nullptr, acc);
}

// ================= Kernel D: W_y = y @ W_w^T + W_b  (f32 -> d_out) =================
__global__ void __launch_bounds__(256)
out_gemm_kernel(const uint16_t* __restrict__ y_bf, const uint16_t* __restrict__ Ww_bf,
                const float* __restrict__ Wb, float* __restrict__ Wy) {
    GEMM_PROLOG
    long rbase = (long)blockIdx.y * 128;
    long cbase = (long)blockIdx.x * 128;   // c in [0, 1024)

    gemm_loop(y_bf, ICN, rbase, Ww_bf, ICN, cbase, ICN,
              sA0, sA1, sB0, sB1, tid, lane, m0, n0, acc);

#pragma unroll
    for (int tm = 0; tm < 2; ++tm)
#pragma unroll
        for (int tn = 0; tn < 4; ++tn) {
            long col = cbase + n0 + tn * 16 + (lane & 15);
            float bv = Wb[col];
            long r0  = rbase + m0 + tm * 16 + 8 * (lane >> 4);
#pragma unroll
            for (int p = 0; p < 8; ++p)
                Wy[(r0 + p) * CC + col] = acc[tm][tn][p] + bv;
        }
}

// ================= Kernel E: per-channel sum / sumsq =================
__global__ void stats_kernel(const float* __restrict__ Wy, float* __restrict__ stats) {
    int c0 = threadIdx.x * 4;
    float s0 = 0, s1 = 0, s2 = 0, s3 = 0, q0 = 0, q1 = 0, q2 = 0, q3 = 0;
    for (long r = blockIdx.x; r < ROWS; r += gridDim.x) {
        float4 d = *(const float4*)(Wy + r * CC + c0);
        s0 += d.x; q0 += d.x * d.x;
        s1 += d.y; q1 += d.y * d.y;
        s2 += d.z; q2 += d.z * d.z;
        s3 += d.w; q3 += d.w * d.w;
    }
    atomicAdd(&stats[c0 + 0], s0);
    atomicAdd(&stats[c0 + 1], s1);
    atomicAdd(&stats[c0 + 2], s2);
    atomicAdd(&stats[c0 + 3], s3);
    atomicAdd(&stats[CC + c0 + 0], q0);
    atomicAdd(&stats[CC + c0 + 1], q1);
    atomicAdd(&stats[CC + c0 + 2], q2);
    atomicAdd(&stats[CC + c0 + 3], q3);
}

// ================= Kernel F: finalize BN scale/shift =================
__global__ void bn_finalize_kernel(const float* __restrict__ stats,
                                   const float* __restrict__ gamma,
                                   const float* __restrict__ beta,
                                   float* __restrict__ ss) {
    int c = blockIdx.x * blockDim.x + threadIdx.x;
    if (c < CC) {
        float inv_n = 1.0f / (float)ROWS;
        float mean  = stats[c] * inv_n;
        float var   = stats[CC + c] * inv_n - mean * mean;
        float sc    = gamma[c] * rsqrtf(var + BN_EPS);
        ss[c]       = sc;
        ss[CC + c]  = beta[c] - mean * sc;
    }
}

// ================= Kernel G: normalize + residual (in place on d_out) =================
__global__ void bn_apply_kernel(float* __restrict__ Wy, const float* __restrict__ v,
                                const float* __restrict__ ss) {
    long idx = ((long)blockIdx.x * blockDim.x + threadIdx.x) * 4;
    int c = (int)(idx & (CC - 1));
    float4 d  = *(float4*)(Wy + idx);
    float4 vv = *(const float4*)(v + idx);
    d.x = d.x * ss[c + 0] + ss[CC + c + 0] + vv.x;
    d.y = d.y * ss[c + 1] + ss[CC + c + 1] + vv.y;
    d.z = d.z * ss[c + 2] + ss[CC + c + 2] + vv.z;
    d.w = d.w * ss[c + 3] + ss[CC + c + 3] + vv.w;
    *(float4*)(Wy + idx) = d;
}

// ================= launch =================
extern "C" void kernel_launch(void* const* d_in, const int* in_sizes, int n_in,
                              void* d_out, int out_size, void* d_ws, size_t ws_size,
                              hipStream_t stream) {
    const float* v     = (const float*)d_in[0];
    const float* gw    = (const float*)d_in[1];
    const float* gb    = (const float*)d_in[2];
    const float* tw    = (const float*)d_in[3];
    const float* tb    = (const float*)d_in[4];
    const float* pw    = (const float*)d_in[5];
    const float* pb    = (const float*)d_in[6];
    const float* Ww    = (const float*)d_in[7];
    const float* Wb    = (const float*)d_in[8];
    const float* gamma = (const float*)d_in[9];
    const float* beta  = (const float*)d_in[10];
    float* out = (float*)d_out;

    char* w = (char*)d_ws;
    auto alloc = [&](size_t bytes) {
        char* p = w;
        w += (bytes + 255) & ~(size_t)255;
        return (void*)p;
    };
    uint16_t* v_bf     = (uint16_t*)alloc((size_t)ROWS * CC * 2);
    uint16_t* theta_bf = (uint16_t*)alloc((size_t)ROWS * ICN * 2);
    uint16_t* phiT     = (uint16_t*)alloc((size_t)BB * ICN * NN * 2);
    uint16_t* gT       = (uint16_t*)alloc((size_t)BB * ICN * NN * 2);
    uint16_t* y_bf     = (uint16_t*)alloc((size_t)ROWS * ICN * 2);
    uint16_t* sT       = (uint16_t*)alloc((size_t)BB * ICN * ICN * 2);
    uint16_t* gw_bf    = (uint16_t*)alloc((size_t)ICN * CC * 2);
    uint16_t* tw_bf    = (uint16_t*)alloc((size_t)ICN * CC * 2);
    uint16_t* pw_bf    = (uint16_t*)alloc((size_t)ICN * CC * 2);
    uint16_t* Ww_bf    = (uint16_t*)alloc((size_t)CC * ICN * 2);
    float*    stats    = (float*)alloc(2 * CC * sizeof(float));
    float*    ss       = (float*)alloc(2 * CC * sizeof(float));

    // 0) convert weights + zero stats; convert v to bf16
    prep_kernel<<<dim3((ICN * CC + 255) / 256), dim3(256), 0, stream>>>(
        gw, tw, pw, Ww, gw_bf, tw_bf, pw_bf, Ww_bf, stats);
    conv_v_kernel<<<dim3((int)(((long)ROWS * CC / 4) / 256)), dim3(256), 0, stream>>>(v, v_bf);

    // A) projections (z: 0=theta, 1=phi, 2=g)
    proj_kernel<<<dim3(ICN / 128, ROWS / 128, 3), dim3(256), 0, stream>>>(
        v_bf, tw_bf, pw_bf, gw_bf, tb, pb, gb, theta_bf, phiT, gT);

    // B) S_b = phi^T g  (per batch)
    s_kernel<<<dim3(ICN / 128, ICN / 128, BB), dim3(256), 0, stream>>>(phiT, gT, sT);

    // C) y = theta @ S / N
    y_kernel<<<dim3(ICN / 128, ROWS / 128), dim3(256), 0, stream>>>(theta_bf, sT, y_bf);

    // D) W_y = y @ W_w^T + W_b -> d_out (pre-BN)
    out_gemm_kernel<<<dim3(CC / 128, ROWS / 128), dim3(256), 0, stream>>>(y_bf, Ww_bf, Wb, out);

    // E) channel stats
    stats_kernel<<<dim3(512), dim3(256), 0, stream>>>(out, stats);

    // F) BN scale/shift
    bn_finalize_kernel<<<dim3(4), dim3(256), 0, stream>>>(stats, gamma, beta, ss);

    // G) normalize + residual
    bn_apply_kernel<<<dim3((int)(((long)ROWS * CC / 4) / 256)), dim3(256), 0, stream>>>(out, v, ss);
}
